// Gate_20298015441099
// MI455X (gfx1250) — compile-verified, matrix operand order checked by reading, not run on hardware
//
#include <hip/hip_runtime.h>
#include <math.h>

typedef __attribute__((ext_vector_type(16))) _Float16 v16h;
typedef __attribute__((ext_vector_type(8)))  _Float16 v8h;
typedef __attribute__((ext_vector_type(8)))  float    v8f;
typedef __attribute__((ext_vector_type(4)))  float    f32x4;

#define N_TOK   16384
#define N_EMBD  4096
#define N_EXP   64
#define TOPK    8
#define WAVES_PER_BLOCK 4
#define TOK_PER_WAVE    32                                  // two 16-row A tiles
#define TOK_PER_BLOCK   (WAVES_PER_BLOCK * TOK_PER_WAVE)    // 128

// ---- fragment loaders -------------------------------------------------------
// CDNA5 16-bit A/B striping: this lane supplies 8 contiguous K at `ka` and 8
// at `ka+16` (ka already accounts for the lane-half split).

// f32 source -> cvt to f16 in-register (used for the streamed x tiles).
__device__ __forceinline__ v16h load_frag_f32cvt(const float* __restrict__ p, int ka) {
    f32x4 c0 = *(const f32x4*)(p + ka);
    f32x4 c1 = *(const f32x4*)(p + ka + 4);
    f32x4 c2 = *(const f32x4*)(p + ka + 16);
    f32x4 c3 = *(const f32x4*)(p + ka + 20);
    v16h h;
    h[0]  = (_Float16)c0[0]; h[1]  = (_Float16)c0[1];
    h[2]  = (_Float16)c0[2]; h[3]  = (_Float16)c0[3];
    h[4]  = (_Float16)c1[0]; h[5]  = (_Float16)c1[1];
    h[6]  = (_Float16)c1[2]; h[7]  = (_Float16)c1[3];
    h[8]  = (_Float16)c2[0]; h[9]  = (_Float16)c2[1];
    h[10] = (_Float16)c2[2]; h[11] = (_Float16)c2[3];
    h[12] = (_Float16)c3[0]; h[13] = (_Float16)c3[1];
    h[14] = (_Float16)c3[2]; h[15] = (_Float16)c3[3];
    return h;
}

// Pre-converted f16 source: two 16B b128 loads, zero VALU (used for W).
__device__ __forceinline__ v16h load_frag_f16(const _Float16* __restrict__ p, int ka) {
    v8h lo = *(const v8h*)(p + ka);
    v8h hi = *(const v8h*)(p + ka + 16);
    return __builtin_shufflevector(lo, hi, 0, 1, 2, 3, 4, 5, 6, 7,
                                           8, 9, 10, 11, 12, 13, 14, 15);
}

// sigmoid via native v_exp_f32 + v_rcp_f32 (~1 ulp; no IEEE div fixup chain)
__device__ __forceinline__ float sigmoidf_(float v) {
    return __builtin_amdgcn_rcpf(1.0f + __expf(-v));
}

#define WMMA_F16(A, B, C) \
    __builtin_amdgcn_wmma_f32_16x16x32_f16(false, (A), false, (B), (short)0, (C), false, false)

// ---- one-shot W f32->f16 pre-pass (1 MB read, 0.5 MB write) -----------------
__global__ __launch_bounds__(256) void w_cvt_kernel(const float* __restrict__ W,
                                                    _Float16* __restrict__ Wh) {
    const size_t i = ((size_t)blockIdx.x * blockDim.x + threadIdx.x) * 8;
    f32x4 a = *(const f32x4*)(W + i);
    f32x4 b = *(const f32x4*)(W + i + 4);
    v8h h;
    h[0] = (_Float16)a[0]; h[1] = (_Float16)a[1];
    h[2] = (_Float16)a[2]; h[3] = (_Float16)a[3];
    h[4] = (_Float16)b[0]; h[5] = (_Float16)b[1];
    h[6] = (_Float16)b[2]; h[7] = (_Float16)b[3];
    *(v8h*)(Wh + i) = h;
}

// ---- main fused gate kernel -------------------------------------------------
template <bool USE_WH>
__global__ __launch_bounds__(128) void moe_gate_kernel(
    const float*    __restrict__ x,     // (N_TOK, N_EMBD) f32
    const float*    __restrict__ W,     // (N_EXP, N_EMBD) f32
    const _Float16* __restrict__ Wh,    // (N_EXP, N_EMBD) f16 (valid if USE_WH)
    const float*    __restrict__ bias,  // (N_EXP,)
    int*   __restrict__ out_idx,        // (N_TOK, TOPK)
    float* __restrict__ out_w)          // (N_TOK, TOPK)
{
    __shared__ float s_scores[WAVES_PER_BLOCK][TOK_PER_WAVE][N_EXP]; // 32 KB
    __shared__ float s_bias[N_EXP];

    const int tid  = threadIdx.x;
    const int wave = tid >> 5;
    const int lane = tid & 31;

    if (tid < N_EXP) s_bias[tid] = bias[tid];

    const int tokBase = blockIdx.x * TOK_PER_BLOCK + wave * TOK_PER_WAVE;
    const int r     = lane & 15;        // row within 16-wide tile
    const int khalf = lane >> 4;        // lane half selects K sub-range

    const float* __restrict__ xrow0 = x + (size_t)(tokBase + r)      * N_EMBD;
    const float* __restrict__ xrow1 = x + (size_t)(tokBase + 16 + r) * N_EMBD;

    const float*    __restrict__ wf0 = W  + (size_t)(r +  0) * N_EMBD;
    const float*    __restrict__ wf1 = W  + (size_t)(r + 16) * N_EMBD;
    const float*    __restrict__ wf2 = W  + (size_t)(r + 32) * N_EMBD;
    const float*    __restrict__ wf3 = W  + (size_t)(r + 48) * N_EMBD;
    const _Float16* __restrict__ wh0 = Wh + (size_t)(r +  0) * N_EMBD;
    const _Float16* __restrict__ wh1 = Wh + (size_t)(r + 16) * N_EMBD;
    const _Float16* __restrict__ wh2 = Wh + (size_t)(r + 32) * N_EMBD;
    const _Float16* __restrict__ wh3 = Wh + (size_t)(r + 48) * N_EMBD;

    // acc[token-tile][expert-tile]
    v8f c00 = {}, c01 = {}, c02 = {}, c03 = {};
    v8f c10 = {}, c11 = {}, c12 = {}, c13 = {};

    for (int k0 = 0; k0 < N_EMBD; k0 += 32) {
        const int ka = k0 + khalf * 8;          // this lane's first K octet

        // branch-free lookahead prefetch for the streamed x rows (wrapped in-row)
        const int kp = (ka + 512) & (N_EMBD - 1);
        __builtin_prefetch(xrow0 + kp, 0, 1);
        __builtin_prefetch(xrow1 + kp, 0, 1);

        v16h a0 = load_frag_f32cvt(xrow0, ka);
        v16h a1 = load_frag_f32cvt(xrow1, ka);
        v16h b0, b1, b2, b3;
        if constexpr (USE_WH) {
            b0 = load_frag_f16(wh0, ka);
            b1 = load_frag_f16(wh1, ka);
            b2 = load_frag_f16(wh2, ka);
            b3 = load_frag_f16(wh3, ka);
        } else {
            b0 = load_frag_f32cvt(wf0, ka);
            b1 = load_frag_f32cvt(wf1, ka);
            b2 = load_frag_f32cvt(wf2, ka);
            b3 = load_frag_f32cvt(wf3, ka);
        }

        // 8 independent WMMAs per k-step; every B fragment feeds two A tiles.
        c00 = WMMA_F16(a0, b0, c00);
        c10 = WMMA_F16(a1, b0, c10);
        c01 = WMMA_F16(a0, b1, c01);
        c11 = WMMA_F16(a1, b1, c11);
        c02 = WMMA_F16(a0, b2, c02);
        c12 = WMMA_F16(a1, b2, c12);
        c03 = WMMA_F16(a0, b3, c03);
        c13 = WMMA_F16(a1, b3, c13);
    }

    // C/D layout: VGPR j holds token row j (lanes 0-15) / row j+8 (lanes 16-31),
    // expert column = lane & 15 within each 16-expert tile.
    const int m_off = khalf * 8;
    const int ecol  = lane & 15;
    float* wbuf = &s_scores[wave][0][0];
#pragma unroll
    for (int j = 0; j < 8; ++j) {
        const int r0 = (m_off + j) * N_EXP;          // token tile 0
        wbuf[r0 +  0 + ecol] = sigmoidf_(c00[j]);
        wbuf[r0 + 16 + ecol] = sigmoidf_(c01[j]);
        wbuf[r0 + 32 + ecol] = sigmoidf_(c02[j]);
        wbuf[r0 + 48 + ecol] = sigmoidf_(c03[j]);
        const int r1 = (16 + m_off + j) * N_EXP;     // token tile 1
        wbuf[r1 +  0 + ecol] = sigmoidf_(c10[j]);
        wbuf[r1 + 16 + ecol] = sigmoidf_(c11[j]);
        wbuf[r1 + 32 + ecol] = sigmoidf_(c12[j]);
        wbuf[r1 + 48 + ecol] = sigmoidf_(c13[j]);
    }

    __syncthreads();

    // Top-8 of 64, one token per lane (all 32 lanes active). Select on
    // (score + bias), keep raw score for the normalized weight; '>' gives the
    // same first-index tie-break as jax.lax.top_k.
    {
        float* sc = &s_scores[wave][lane][0];
        int   ids[TOPK];
        float wts[TOPK];
        float wsum = 0.0f;
#pragma unroll
        for (int i = 0; i < TOPK; ++i) {
            float best = -INFINITY;
            int   bi   = 0;
            for (int e = 0; e < N_EXP; ++e) {
                float v = sc[e] + s_bias[e];
                if (v > best) { best = v; bi = e; }
            }
            float raw = sc[bi];
            sc[bi] = -INFINITY;
            ids[i] = bi;
            wts[i] = raw;
            wsum  += raw;
        }
        const float inv = __builtin_amdgcn_rcpf(wsum);   // ROUTE_SCALE == 1.0
        const int token = tokBase + lane;
#pragma unroll
        for (int i = 0; i < TOPK; ++i) {
            out_idx[(size_t)token * TOPK + i] = ids[i];
            out_w  [(size_t)token * TOPK + i] = wts[i] * inv;
        }
    }
}

extern "C" void kernel_launch(void* const* d_in, const int* in_sizes, int n_in,
                              void* d_out, int out_size, void* d_ws, size_t ws_size,
                              hipStream_t stream) {
    const float* x    = (const float*)d_in[0];
    const float* W    = (const float*)d_in[1];
    const float* bias = (const float*)d_in[2];

    // d_out = [topk_idx (int32, N_TOK*TOPK)] ++ [weights (f32, N_TOK*TOPK)]
    int*   out_idx = (int*)d_out;
    float* out_w   = (float*)d_out + (size_t)N_TOK * TOPK;

    const dim3 grid(N_TOK / TOK_PER_BLOCK);   // 128 blocks
    const dim3 block(WAVES_PER_BLOCK * 32);   // 128 threads = 4 wave32s

    const size_t whBytes = (size_t)N_EXP * N_EMBD * sizeof(_Float16);  // 512 KB
    if (ws_size >= whBytes) {
        _Float16* Wh = (_Float16*)d_ws;
        // one-shot W f32 -> f16 (re-run every call: deterministic, negligible)
        hipLaunchKernelGGL(w_cvt_kernel,
                           dim3((N_EXP * N_EMBD / 8) / 256), dim3(256), 0, stream,
                           W, Wh);
        hipLaunchKernelGGL((moe_gate_kernel<true>), grid, block, 0, stream,
                           x, W, Wh, bias, out_idx, out_w);
    } else {
        hipLaunchKernelGGL((moe_gate_kernel<false>), grid, block, 0, stream,
                           x, W, (const _Float16*)nullptr, bias, out_idx, out_w);
    }
}
